// ConvIntrinsic_17102559772776
// MI455X (gfx1250) — matrix operand order repaired and checked by reading, not compiled, not run on hardware
//
#include <hip/hip_runtime.h>

#define N_PTS 50000
#define RDIM 5
#define ADIM 8
#define FDIM 32
#define TDIM 32
#define KREAL 1312         // R*A*F + F  (neighbor K + center K)
#define KPAD  1344         // padded to 42 chunks of 32 (even trip count)
#define CDIM 256           // O*T
#define KC   42            // KPAD / 32  (even -> clean unroll-by-2 pipeline)
#define MTILE 32           // 2 row tiles of 16 per block
#define LDSTRIDE 1352      // KPAD + 8: 16B-aligned rows, gcd(stride_dwords,64)=4
#define LDSBYTES (MTILE * LDSTRIDE * 2)

typedef __attribute__((ext_vector_type(16))) __bf16 v16bf;
typedef __attribute__((ext_vector_type(8)))  __bf16 v8bf;
typedef __attribute__((ext_vector_type(8)))  float  v8f;

// ---------------------------------------------------------------------------
// Fold Gaussian prior + 8 rotations + center weights into one bf16 weight
// matrix, K-major per output column:  WT[c][k], c = o*32+t, k in [0,1344)
//   k < 1280        : k=(x*8+y)*32+f -> sum_{r,a} kernel[r,a,x,y]*Wn[t,r,(a+o)%8,f]
//   1280 <= k <1312 : W_self[t, k-1280]
//   k >= 1312       : 0  (K padding; d_ws is poisoned, must be written)
// ---------------------------------------------------------------------------
__global__ __launch_bounds__(256) void fold_weights_kernel(
    const float* __restrict__ Wn, const float* __restrict__ Ws,
    const float* __restrict__ gk, __bf16* __restrict__ WT) {
  const int c = blockIdx.x;          // 0..255
  const int o = c >> 5, t = c & 31;
  for (int k = threadIdx.x; k < KPAD; k += blockDim.x) {
    float acc;
    if (k < RDIM * ADIM * FDIM) {
      const int x = k >> 8;
      const int y = (k >> 5) & 7;
      const int f = k & 31;
      acc = 0.f;
#pragma unroll
      for (int r = 0; r < RDIM; ++r)
#pragma unroll
        for (int a = 0; a < ADIM; ++a) {
          const int arot = (a + o) & 7;
          acc += gk[((r * ADIM + a) * RDIM + x) * ADIM + y] *
                 Wn[((t * RDIM + r) * ADIM + arot) * FDIM + f];
        }
    } else if (k < KREAL) {
      acc = Ws[t * FDIM + (k - RDIM * ADIM * FDIM)];
    } else {
      acc = 0.f;
    }
    WT[(size_t)c * KPAD + k] = (__bf16)acc;
  }
}

// ---------------------------------------------------------------------------
// Main kernel: gather+interp (fp32) -> bf16 LDS tile [32 x 1344], then
// WMMA GEMM [32 x 1344] x [1344 x 256] with fused bias+relu.
// Each of the 8 waves owns a 2x2 block of 16x16 C tiles:
//   row tiles {0,1}, col tiles {2*wave, 2*wave+1}
// -> per 32-K chunk: 2 A frags (4 ds_b128) + 2 B frags (4 global_b128)
//    feed 4 v_wmma  (1 vmem b128 per wmma, 1 ds b128 per wmma).
// ---------------------------------------------------------------------------
__global__ __launch_bounds__(256) void conv_main_kernel(
    const float* __restrict__ mesh, const float* __restrict__ bary,
    const float* __restrict__ bias, const __bf16* __restrict__ WT,
    float* __restrict__ out) {
  extern __shared__ char smem_raw[];
  __bf16* Alds = (__bf16*)smem_raw;          // [MTILE][LDSTRIDE]

  const int tid  = threadIdx.x;
  const int wave = tid >> 5;                 // 0..7
  const int lane = tid & 31;
  const int n0   = blockIdx.x * MTILE;

  // ---- stage 1: barycentric interpolation into LDS (lane = feature f) ----
  for (int pi = 0; pi < 4; ++pi) {
    const int p = wave * 4 + pi;             // local row 0..31
    const int n = n0 + p;
    __bf16* arow_st = Alds + (size_t)p * LDSTRIDE;
    arow_st[KREAL + lane] = (__bf16)0.f;     // K padding [1312,1344)
    if (n < N_PTS) {
      arow_st[RDIM * ADIM * FDIM + lane] = (__bf16)mesh[(size_t)n * FDIM + lane];
      const float* bc = bary + (size_t)n * (RDIM * ADIM * 3 * 2);
#pragma unroll 4
      for (int cell = 0; cell < RDIM * ADIM; ++cell) {
        const float* cb = bc + cell * 6;     // wave-uniform address -> SMEM loads
        const float i0 = cb[0], w0 = cb[1];
        const float i1 = cb[2], w1 = cb[3];
        const float i2 = cb[4], w2 = cb[5];
        const int j0 = (int)i0, j1 = (int)i1, j2 = (int)i2;
        const float g = w0 * mesh[(size_t)j0 * FDIM + lane]
                      + w1 * mesh[(size_t)j1 * FDIM + lane]
                      + w2 * mesh[(size_t)j2 * FDIM + lane];
        arow_st[cell * FDIM + lane] = (__bf16)g;
      }
    } else {
      arow_st[RDIM * ADIM * FDIM + lane] = (__bf16)0.f;
      for (int cell = 0; cell < RDIM * ADIM; ++cell)
        arow_st[cell * FDIM + lane] = (__bf16)0.f;
    }
  }
  __syncthreads();

  // ---- stage 2: WMMA GEMM, 2x2 register blocking per wave ----
  const int m   = lane & 15;
  const int h   = lane >> 4;
  const int ct0 = wave * 2;
  v8f acc00 = {}, acc01 = {}, acc10 = {}, acc11 = {};
  const __bf16* arow0 = Alds + (size_t)m * LDSTRIDE;         // row tile 0
  const __bf16* arow1 = Alds + (size_t)(16 + m) * LDSTRIDE;  // row tile 1
  const __bf16* b0row = WT + (size_t)(ct0 * 16 + m) * KPAD;
  const __bf16* b1row = WT + (size_t)(ct0 * 16 + 16 + m) * KPAD;

  for (int kc = 0; kc < KC; ++kc) {
    const int kb = kc * 32;
    // A fragments (contiguous K order, same as B): lane m/h holds row m,
    // K = {16h..16h+7} and {16h+8..16h+15}  -> two 16B LDS loads each
    const v8bf lo0 = *(const v8bf*)(arow0 + kb + 16 * h);
    const v8bf hi0 = *(const v8bf*)(arow0 + kb + 16 * h + 8);
    const v8bf lo1 = *(const v8bf*)(arow1 + kb + 16 * h);
    const v8bf hi1 = *(const v8bf*)(arow1 + kb + 16 * h + 8);
    v16bf a0, a1;
#pragma unroll
    for (int j = 0; j < 8; ++j) {
      a0[j] = lo0[j]; a0[j + 8] = hi0[j];
      a1[j] = lo1[j]; a1[j + 8] = hi1[j];
    }
    // B fragments (32x16): lane holds column m, K = 16h..16h+15 contiguous
    const v16bf b0 = *(const v16bf*)(b0row + kb + 16 * h);
    const v16bf b1 = *(const v16bf*)(b1row + kb + 16 * h);
    acc00 = __builtin_amdgcn_wmma_f32_16x16x32_bf16(false, a0, false, b0,
                                                    (short)0, acc00, false, false);
    acc01 = __builtin_amdgcn_wmma_f32_16x16x32_bf16(false, a0, false, b1,
                                                    (short)0, acc01, false, false);
    acc10 = __builtin_amdgcn_wmma_f32_16x16x32_bf16(false, a1, false, b0,
                                                    (short)0, acc10, false, false);
    acc11 = __builtin_amdgcn_wmma_f32_16x16x32_bf16(false, a1, false, b1,
                                                    (short)0, acc11, false, false);
  }

  // ---- stage 3: bias + relu + store (C/D layout: VGPR v -> M = v + 8h) ----
  const int col0 = ct0 * 16 + m;
  const int col1 = col0 + 16;
  const float bv0 = bias[col0 & 31];
  const float bv1 = bias[col1 & 31];
#pragma unroll
  for (int v = 0; v < 8; ++v) {
    const int row = v + 8 * h;
    const int nA = n0 + row;          // row tile 0
    const int nB = n0 + 16 + row;     // row tile 1
    if (nA < N_PTS) {
      const float x00 = acc00[v] + bv0;
      const float x01 = acc01[v] + bv1;
      out[(size_t)nA * CDIM + col0] = x00 > 0.f ? x00 : 0.f;
      out[(size_t)nA * CDIM + col1] = x01 > 0.f ? x01 : 0.f;
    }
    if (nB < N_PTS) {
      const float x10 = acc10[v] + bv0;
      const float x11 = acc11[v] + bv1;
      out[(size_t)nB * CDIM + col0] = x10 > 0.f ? x10 : 0.f;
      out[(size_t)nB * CDIM + col1] = x11 > 0.f ? x11 : 0.f;
    }
  }
}

extern "C" void kernel_launch(void* const* d_in, const int* in_sizes, int n_in,
                              void* d_out, int out_size, void* d_ws, size_t ws_size,
                              hipStream_t stream) {
  const float* mesh = (const float*)d_in[0];   // [N, F]
  const float* bary = (const float*)d_in[1];   // [N, R, A, 3, 2]
  const float* Ws   = (const float*)d_in[2];   // [T, 1, F]
  const float* Wn   = (const float*)d_in[3];   // [T, R, A, F]
  const float* bias = (const float*)d_in[4];   // [1, T]
  const float* gk   = (const float*)d_in[5];   // [R, A, R, A]
  float* out = (float*)d_out;                  // [N, O, T]

  __bf16* WT = (__bf16*)d_ws;                  // 256 * 1344 * 2 B = 688,128 B

  fold_weights_kernel<<<CDIM, 256, 0, stream>>>(Wn, Ws, gk, WT);

  const int nblk = (N_PTS + MTILE - 1) / MTILE;
  conv_main_kernel<<<nblk, 256, LDSBYTES, stream>>>(mesh, bary, bias, WT, out);
}
